// MultiAxisAttention_86964497809634
// MI455X (gfx1250) — matrix operand order repaired and checked by reading, not compiled, not run on hardware
//
#include <hip/hip_runtime.h>

typedef __attribute__((ext_vector_type(16))) __bf16 v16bf;
typedef __attribute__((ext_vector_type(8)))  float  v8f;

#define WINSZ 49
#define NHEAD 16
#define DHEAD 32
#define DIM   512
#define NQKV  1536
static constexpr float ATT_SCALE = 0.17677669529663687f; // 32^-0.5

__device__ __forceinline__ unsigned short f2bf(float f) {
  unsigned int u = __float_as_uint(f);
  u += 0x7fffu + ((u >> 16) & 1u);            // round-to-nearest-even
  return (unsigned short)(u >> 16);
}

// Build a 16xbf16 fragment from two 16-byte chunks (LDS or global).
__device__ __forceinline__ v16bf ld_frag(const unsigned short* p0, const unsigned short* p1) {
  union { v16bf v; uint4 u[2]; } t;
  t.u[0] = *(const uint4*)p0;
  t.u[1] = *(const uint4*)p1;
  return t.v;
}

// ---- prep: bf16-transpose weights, materialize per-head bias matrix ----
__global__ void prep_kernel(const float* __restrict__ wqkv, const float* __restrict__ wout,
                            const float* __restrict__ btab,
                            unsigned short* __restrict__ wqkvT,  // [1536][512] bf16
                            unsigned short* __restrict__ woutT,  // [512][512]  bf16
                            float* __restrict__ biasm) {         // [16][49][49] f32
  int i = blockIdx.x * 256 + threadIdx.x;
  const int NW1 = NQKV * DIM, NW2 = DIM * DIM, NB = NHEAD * WINSZ * WINSZ;
  if (i < NW1) {
    int n = i >> 9, k = i & 511;
    wqkvT[i] = f2bf(wqkv[k * NQKV + n]);
  } else if (i < NW1 + NW2) {
    int j = i - NW1;
    int n = j >> 9, k = j & 511;
    woutT[j] = f2bf(wout[k * DIM + n]);
  } else if (i < NW1 + NW2 + NB) {
    int j = i - NW1 - NW2;
    int h = j / (WINSZ * WINSZ), rem = j % (WINSZ * WINSZ);
    int ii = rem / WINSZ, jj = rem % WINSZ;
    int rel = (ii / 7 - jj / 7 + 6) * 13 + (ii % 7 - jj % 7 + 6);
    biasm[j] = btab[rel * NHEAD + h];
  }
}

// ---- fused windowed attention: one workgroup per window ----
__global__ __launch_bounds__(256)
void attn_kernel(const float* __restrict__ x, float* __restrict__ out,
                 const unsigned short* __restrict__ wqkvT,
                 const unsigned short* __restrict__ woutT,
                 const float* __restrict__ biasm) {
  extern __shared__ unsigned short smem[];
  unsigned short* XS = smem;           // [64][512] bf16 x ; phase2: per-wave attn scratch
  unsigned short* Q  = smem + 32768;   // [16][64][32] q ; phase2 output o overwrites
  unsigned short* K  = smem + 65536;   // [16][64][32] k row-major
  unsigned short* V  = smem + 98304;   // [16][32][64] v transposed

  const int tid  = threadIdx.x;
  const int wid  = tid >> 5;
  const int lane = tid & 31;
  const int lh   = lane >> 4;          // lane half
  const int nlo  = lane & 15;
  const int win  = blockIdx.x;
  const float* xw = x + (size_t)win * WINSZ * DIM;

  // Phase 0: x window f32 -> bf16 XS, rows 49..63 zero-padded
  for (int g = tid; g < 64 * 128; g += 256) {
    int row  = g >> 7;
    int col4 = (g & 127) << 2;
    float4 f = make_float4(0.f, 0.f, 0.f, 0.f);
    if (row < WINSZ) f = *(const float4*)(xw + row * DIM + col4);
    uint2 p;
    p.x = (unsigned)f2bf(f.x) | ((unsigned)f2bf(f.y) << 16);
    p.y = (unsigned)f2bf(f.z) | ((unsigned)f2bf(f.w) << 16);
    *(uint2*)(XS + row * DIM + col4) = p;
  }
  __syncthreads();

  // Phase 1: QKV GEMM  [64,512] @ [512,1536]; wave owns 12 N-tiles x 4 M-tiles
  for (int j = 0; j < 12; ++j) {
    int nt = wid * 12 + j;
    int n0 = nt << 4;
    v8f acc[4];
    #pragma unroll
    for (int mt = 0; mt < 4; ++mt) acc[mt] = (v8f){};
    const unsigned short* brow = wqkvT + (size_t)(n0 + nlo) * DIM;
    for (int ks = 0; ks < 16; ++ks) {
      const unsigned short* bp = brow + ks * 32 + lh * 8;
      v16bf b = ld_frag(bp, bp + 16);
      #pragma unroll
      for (int mt = 0; mt < 4; ++mt) {
        const unsigned short* ap = XS + (mt * 16 + nlo) * DIM + ks * 32 + lh * 8;
        v16bf a = ld_frag(ap, ap + 16);
        acc[mt] = __builtin_amdgcn_wmma_f32_16x16x32_bf16(false, a, false, b,
                                                          (short)0, acc[mt], false, false);
      }
    }
    int s  = n0 >> 9;          // 0=q 1=k 2=v
    int h  = (n0 & 511) >> 5;
    int d0 = ((n0 & 31)) + nlo;
    #pragma unroll
    for (int mt = 0; mt < 4; ++mt) {
      if (s < 2) {
        unsigned short* base = (s == 0 ? Q : K) + h * 2048 + d0;
        #pragma unroll
        for (int r = 0; r < 8; ++r) {
          int m = mt * 16 + r + lh * 8;
          base[m * 32] = f2bf(acc[mt][r]);
        }
      } else { // v transposed: 8 consecutive rows -> one b128 store
        union { uint4 u; unsigned short s16[8]; } pk;
        #pragma unroll
        for (int r = 0; r < 8; ++r) pk.s16[r] = f2bf(acc[mt][r]);
        *(uint4*)(V + h * 2048 + d0 * 64 + mt * 16 + lh * 8) = pk.u;
      }
    }
  }
  __syncthreads();

  // Phase 2: attention, wave handles heads wid and wid+8
  unsigned short* ATT = XS + wid * 4096;   // per-wave [64][64] bf16 scratch
  for (int hi = 0; hi < 2; ++hi) {
    int h = wid + hi * 8;
    const unsigned short* qh = Q + h * 2048;
    const unsigned short* kh = K + h * 2048;
    const unsigned short* vh = V + h * 2048;
    const float* bh = biasm + h * (WINSZ * WINSZ);

    for (int tm = 0; tm < 4; ++tm) {
      const unsigned short* ap = qh + (tm * 16 + nlo) * 32 + lh * 8;
      v16bf a = ld_frag(ap, ap + 16);
      v8f c[4];
      #pragma unroll
      for (int tn = 0; tn < 4; ++tn) {
        const unsigned short* bp = kh + (tn * 16 + nlo) * 32 + lh * 8;
        v16bf b = ld_frag(bp, bp + 16);
        v8f z = (v8f){};
        c[tn] = __builtin_amdgcn_wmma_f32_16x16x32_bf16(false, a, false, b,
                                                        (short)0, z, false, false);
      }
      // scale + relative-position bias + column mask
      #pragma unroll
      for (int tn = 0; tn < 4; ++tn) {
        int n = tn * 16 + nlo;
        #pragma unroll
        for (int r = 0; r < 8; ++r) {
          int m = tm * 16 + r + lh * 8;
          float v = -1e30f;
          if (m < WINSZ && n < WINSZ) v = c[tn][r] * ATT_SCALE + bh[m * WINSZ + n];
          c[tn][r] = v;
        }
      }
      // row softmax: in-lane over 4 N-tiles, then 16-lane shfl_xor reduction
      #pragma unroll
      for (int r = 0; r < 8; ++r) {
        float mx = fmaxf(fmaxf(c[0][r], c[1][r]), fmaxf(c[2][r], c[3][r]));
        #pragma unroll
        for (int off = 1; off < 16; off <<= 1) mx = fmaxf(mx, __shfl_xor(mx, off, 32));
        float e0 = __expf(c[0][r] - mx), e1 = __expf(c[1][r] - mx);
        float e2 = __expf(c[2][r] - mx), e3 = __expf(c[3][r] - mx);
        float sm = e0 + e1 + e2 + e3;
        #pragma unroll
        for (int off = 1; off < 16; off <<= 1) sm += __shfl_xor(sm, off, 32);
        float inv = 1.0f / sm;
        c[0][r] = e0 * inv; c[1][r] = e1 * inv; c[2][r] = e2 * inv; c[3][r] = e3 * inv;
      }
      #pragma unroll
      for (int tn = 0; tn < 4; ++tn) {
        #pragma unroll
        for (int r = 0; r < 8; ++r) {
          int m = tm * 16 + r + lh * 8;
          ATT[m * 64 + tn * 16 + nlo] = f2bf(c[tn][r]);
        }
      }
    }
    // out_head = attn[64,64] @ v[64,32]; overwrite dead q slot with o
    unsigned short* oh = Q + h * 2048;
    #pragma unroll
    for (int tm = 0; tm < 4; ++tm) {
      #pragma unroll
      for (int dn = 0; dn < 2; ++dn) {
        v8f acc = (v8f){};
        #pragma unroll
        for (int ks = 0; ks < 2; ++ks) {
          const unsigned short* ap2 = ATT + (tm * 16 + nlo) * 64 + ks * 32 + lh * 8;
          v16bf a2 = ld_frag(ap2, ap2 + 16);
          const unsigned short* bp2 = vh + (dn * 16 + nlo) * 64 + ks * 32 + lh * 8;
          v16bf b2 = ld_frag(bp2, bp2 + 16);
          acc = __builtin_amdgcn_wmma_f32_16x16x32_bf16(false, a2, false, b2,
                                                        (short)0, acc, false, false);
        }
        #pragma unroll
        for (int r = 0; r < 8; ++r) {
          int m = tm * 16 + r + lh * 8;
          oh[m * 32 + dn * 16 + nlo] = f2bf(acc[r]);
        }
      }
    }
  }
  __syncthreads();

  // Phase 3: out-proj  o[64,512] @ [512,512] -> global f32; wave owns 4 N-tiles
  for (int j = 0; j < 4; ++j) {
    int nt = wid * 4 + j;
    v8f acc[4];
    #pragma unroll
    for (int mt = 0; mt < 4; ++mt) acc[mt] = (v8f){};
    const unsigned short* brow = woutT + (size_t)(nt * 16 + nlo) * DIM;
    for (int ks = 0; ks < 16; ++ks) {  // K group ks == head index in o layout
      const unsigned short* bp = brow + ks * 32 + lh * 8;
      v16bf b = ld_frag(bp, bp + 16);
      #pragma unroll
      for (int mt = 0; mt < 4; ++mt) {
        const unsigned short* ap = Q + ks * 2048 + (mt * 16 + nlo) * 32 + lh * 8;
        v16bf a = ld_frag(ap, ap + 16);
        acc[mt] = __builtin_amdgcn_wmma_f32_16x16x32_bf16(false, a, false, b,
                                                          (short)0, acc[mt], false, false);
      }
    }
    float* ocol = out + (size_t)win * WINSZ * DIM + nt * 16 + nlo;
    #pragma unroll
    for (int mt = 0; mt < 4; ++mt) {
      #pragma unroll
      for (int r = 0; r < 8; ++r) {
        int m = mt * 16 + r + lh * 8;
        if (m < WINSZ) ocol[(size_t)m * DIM] = acc[mt][r];
      }
    }
  }
}

extern "C" void kernel_launch(void* const* d_in, const int* in_sizes, int n_in,
                              void* d_out, int out_size, void* d_ws, size_t ws_size,
                              hipStream_t stream) {
  const float* x    = (const float*)d_in[0];
  const float* wqkv = (const float*)d_in[1];
  const float* btab = (const float*)d_in[2];
  const float* wout = (const float*)d_in[3];

  unsigned short* wqkvT = (unsigned short*)d_ws;                 // 1.5 MB
  unsigned short* woutT = wqkvT + NQKV * DIM;                    // 0.5 MB
  float*          biasm = (float*)(woutT + DIM * DIM);           // 150 KB

  int prep_total = NQKV * DIM + DIM * DIM + NHEAD * WINSZ * WINSZ;
  prep_kernel<<<(prep_total + 255) / 256, 256, 0, stream>>>(wqkv, wout, btab,
                                                            wqkvT, woutT, biasm);

  int nwin = in_sizes[0] / (WINSZ * DIM);                        // 16*196 = 3136
  attn_kernel<<<nwin, 256, 262144, stream>>>(x, (float*)d_out, wqkvT, woutT, biasm);
}
